// QAPR_Net_12266426597557
// MI455X (gfx1250) — matrix operand order repaired and checked by reading, not compiled, not run on hardware
//
#include <hip/hip_runtime.h>
#include <math.h>

// ---------------- problem constants ----------------
#define N_WAY 5
#define K_SHOT 5
#define PRJ 14
#define DD 64
#define HW 100
#define QQ 1024
#define NS (N_WAY*K_SHOT*PRJ)          // 350
#define FLEN (DD*HW)                   // 6400
#define NSAMP (NS+QQ)                  // 1374
#define KT1 (FLEN/32)                  // 200 k-tiles for GEMM1
#define KSPLIT 8                       // K split for GEMM1 (25 k-tiles each)
#define NKT (KT1/KSPLIT)               // 25
#define NPAD 80                        // 70 -> 80 (5 n-tiles)
#define COL2 (PRJ*DD)                  // 896 = 56 n-tiles
#define NT2 56
#define KPAD2 128                      // 100 -> 128 (4 k-tiles)

typedef __attribute__((ext_vector_type(16))) __bf16 v16bf;
typedef __attribute__((ext_vector_type(8)))  float  v8f;

union BF16x16 { v16bf v; uint4 q[2]; unsigned short s[16]; };

__device__ __forceinline__ unsigned short f2bf(float x) {
  union { float f; unsigned int u; } c; c.f = x;
  unsigned int u = c.u;
  unsigned int r = u + 0x7fffu + ((u >> 16) & 1u);
  return (unsigned short)(r >> 16);
}

__device__ __forceinline__ float wave_sum(float v) {
  #pragma unroll
  for (int off = 16; off > 0; off >>= 1) v += __shfl_xor(v, off, 32);
  return v;
}

// ---------------- prep kernels ----------------
__global__ void init_acc_kernel(float* acc) {
  if (threadIdx.x < 2) acc[threadIdx.x] = 0.0f;
}

// per-(sample, hw) inverse L2 norm over D
__global__ void p0_invnorm(const float* __restrict__ feat, float* __restrict__ inv_s) {
  int tid = blockIdx.x * 256 + threadIdx.x;
  if (tid >= NSAMP * HW) return;
  int s = tid / HW, hw = tid % HW;
  const float* f = feat + (size_t)s * FLEN + hw;
  float acc = 0.f;
  #pragma unroll 8
  for (int d = 0; d < DD; ++d) { float x = f[d * HW]; acc = fmaf(x, x, acc); }
  inv_s[tid] = 1.0f / fmaxf(sqrtf(acc), 1e-12f);
}

// view_proto[n][v][d][hw] = mean_k normalized support
__global__ void p1_viewproto(const float* __restrict__ feat, const float* __restrict__ inv_s,
                             float* __restrict__ vp) {
  int tid = blockIdx.x * 256 + threadIdx.x;
  if (tid >= N_WAY * PRJ * DD * HW) return;
  int hw = tid % HW;
  int d  = (tid / HW) % DD;
  int v  = (tid / FLEN) % PRJ;
  int n  = tid / (FLEN * PRJ);
  float acc = 0.f;
  #pragma unroll
  for (int k = 0; k < K_SHOT; ++k) {
    int s = n * (K_SHOT * PRJ) + k * PRJ + v;
    acc = fmaf(feat[(size_t)s * FLEN + d * HW + hw], inv_s[s * HW + hw], acc);
  }
  vp[tid] = acc * 0.2f;
}

// per-(n,v) inverse norm of flattened view_proto
__global__ void p2_invvf(const float* __restrict__ vp, float* __restrict__ inv_vf) {
  __shared__ float red[256];
  int b = blockIdx.x, t = threadIdx.x;
  const float* row = vp + (size_t)b * FLEN;
  float s = 0.f;
  for (int i = t; i < FLEN; i += 256) { float x = row[i]; s = fmaf(x, x, s); }
  red[t] = s; __syncthreads();
  for (int o = 128; o > 0; o >>= 1) { if (t < o) red[t] += red[t + o]; __syncthreads(); }
  if (t == 0) inv_vf[b] = 1.0f / fmaxf(sqrtf(red[0]), 1e-12f);
}

// colsumV[nv][hw] = sum_d vp
__global__ void p2b_colsum(const float* __restrict__ vp, float* __restrict__ colsumV) {
  int tid = blockIdx.x * 256 + threadIdx.x;
  if (tid >= N_WAY * PRJ * HW) return;
  int nv = tid / HW, hw = tid % HW;
  float s = 0.f;
  #pragma unroll 8
  for (int d = 0; d < DD; ++d) s += vp[(size_t)nv * FLEN + d * HW + hw];
  colsumV[tid] = s;
}

// base[n][d] = mean over (v,hw)
__global__ void p6_base(const float* __restrict__ vp, float* __restrict__ base) {
  int tid = blockIdx.x * 256 + threadIdx.x;
  if (tid >= N_WAY * DD) return;
  int n = tid / DD, d = tid % DD;
  float s = 0.f;
  for (int v = 0; v < PRJ; ++v) {
    const float* p = vp + ((size_t)(n * PRJ + v) * DD + d) * HW;
    for (int hw = 0; hw < HW; ++hw) s += p[hw];
  }
  base[tid] = s * (1.0f / (PRJ * HW));
}

// pack B1 for GEMM1 into WMMA B layout: b1p[nt][kt][lane][e]
__global__ void p3_packB1(const float* __restrict__ vp, const float* __restrict__ inv_vf,
                          unsigned short* __restrict__ b1p) {
  int tid = blockIdx.x * 256 + threadIdx.x;
  if (tid >= 5 * KT1 * 32 * 16) return;
  int e = tid & 15, lane = (tid >> 4) & 31;
  int rem = tid >> 9;
  int kt = rem % KT1, nt = rem / KT1;
  int nv = nt * 16 + (lane & 15);
  int k  = kt * 32 + (lane >> 4) * 8 + (e & 7) + (e >> 3) * 16;
  unsigned short o = 0;
  if (nv < N_WAY * PRJ) o = f2bf(vp[(size_t)nv * FLEN + k] * inv_vf[nv]);
  b1p[tid] = o;
}

// query_vec[q][d] = mean_hw normalized query
__global__ void p4_qvec(const float* __restrict__ feat, const float* __restrict__ inv_s,
                        float* __restrict__ qvec) {
  int tid = blockIdx.x * 256 + threadIdx.x;
  if (tid >= QQ * DD) return;
  int q = tid >> 6, d = tid & 63;
  int s = NS + q;
  const float* f = feat + (size_t)s * FLEN + d * HW;
  const float* iv = inv_s + s * HW;
  float acc = 0.f;
  #pragma unroll 4
  for (int hw = 0; hw < HW; ++hw) acc = fmaf(f[hw], iv[hw], acc);
  qvec[tid] = acc * 0.01f;
}

// q_bf[q][f] = normalized query in bf16 (GEMM1 A operand, row-major)
__global__ void p4b_qbf(const float* __restrict__ feat, const float* __restrict__ inv_s,
                        unsigned short* __restrict__ qbf) {
  int tid = blockIdx.x * 256 + threadIdx.x;
  if (tid >= QQ * FLEN) return;
  int q = tid / FLEN, r = tid % FLEN;
  int hw = r % HW;
  int s = NS + q;
  qbf[tid] = f2bf(feat[(size_t)s * FLEN + r] * inv_s[s * HW + hw]);
}

// pack B2 for GEMM2: b2p[n][nt][kt][lane][e]; B2[k=hw][col=v*64+d] = vp[n][v][d][hw]
__global__ void p5_packB2(const float* __restrict__ vp, unsigned short* __restrict__ b2p) {
  int tid = blockIdx.x * 256 + threadIdx.x;
  if (tid >= N_WAY * NT2 * 4 * 32 * 16) return;
  int e = tid & 15, lane = (tid >> 4) & 31;
  int rem = tid >> 9;
  int kt = rem & 3; rem >>= 2;
  int nt = rem % NT2, n = rem / NT2;
  int col = nt * 16 + (lane & 15);
  int v = col >> 6, d = col & 63;
  int k = kt * 32 + (lane >> 4) * 8 + (e & 7) + (e >> 3) * 16;
  unsigned short o = 0;
  if (k < HW) o = f2bf(vp[((size_t)(n * PRJ + v) * DD + d) * HW + k]);
  b2p[tid] = o;
}

// ---------- WMMA GEMM1 (K-split, double-buffered, running pointers) ----------
// sim_part[ks][1024][80]
__global__ __launch_bounds__(32) void g1_gemm(const unsigned short* __restrict__ qbf,
                                              const unsigned short* __restrict__ b1p,
                                              float* __restrict__ sim_part) {
  const int lane = threadIdx.x;
  const int qbase = blockIdx.x * 16;
  const int ks = blockIdx.y;
  const int m = lane & 15, half = lane >> 4;
  const int kt0 = ks * NKT;
  v8f c[5] = {};

  // running pointers: A advances 32 elems (64 B) per tile, B 512 elems (1 KB) per tile
  const unsigned short* ap = qbf + (size_t)(qbase + m) * FLEN + kt0 * 32 + half * 8;
  const unsigned short* bp = b1p + (size_t)kt0 * 512 + lane * 16;

  auto LOADT = [&](BF16x16& a, BF16x16 (&b)[5]) {
    a.q[0] = *(const uint4*)(ap);
    a.q[1] = *(const uint4*)(ap + 16);
    #pragma unroll
    for (int nt = 0; nt < 5; ++nt) {
      // nt stride = KT1*512 elems = 204800 B (fits the 24-bit signed IOFFSET)
      const unsigned short* p = bp + nt * (KT1 * 512);
      b[nt].q[0] = *(const uint4*)(p);
      b[nt].q[1] = *(const uint4*)(p + 8);
    }
    __builtin_prefetch((const void*)(ap + 4 * 32), 0, 3);  // WGP-scope, 4 tiles ahead
    ap += 32;
    bp += 512;
  };
  auto WMMA5 = [&](BF16x16& a, BF16x16 (&b)[5]) {
    #pragma unroll
    for (int nt = 0; nt < 5; ++nt)
      c[nt] = __builtin_amdgcn_wmma_f32_16x16x32_bf16(false, a.v, false, b[nt].v,
                                                      (short)0, c[nt], false, false);
  };

  BF16x16 aA, aB;
  BF16x16 bA[5], bB[5];
  LOADT(aA, bA);                        // kt0
  int i = 0;
  for (; i + 2 <= NKT; i += 2) {
    LOADT(aB, bB);                      // kt0+i+1 in flight
    WMMA5(aA, bA);                      // consume kt0+i
    if (i + 2 < NKT) LOADT(aA, bA);     // kt0+i+2 in flight
    WMMA5(aB, bB);                      // consume kt0+i+1
  }
  if (i < NKT) WMMA5(aA, bA);           // odd tail (already resident)

  #pragma unroll
  for (int nt = 0; nt < 5; ++nt)
    #pragma unroll
    for (int r = 0; r < 8; ++r)
      sim_part[((size_t)ks * QQ + qbase + r + 8 * half) * NPAD + nt * 16 + m] = c[nt][r];
}

// ---------------- softmax p_w + spatial attn (reduces K-split partials) ----------------
__global__ __launch_bounds__(128) void s1_softmax(const float* __restrict__ sim_part,
                                                  const float* __restrict__ colsumV,
                                                  const float* __restrict__ temp,
                                                  float* __restrict__ p_w,
                                                  unsigned short* __restrict__ attn_bf) {
  int q = blockIdx.x, n = blockIdx.y, t = threadIdx.x;
  float invt = 1.0f / (temp[0] + 1e-6f);
  float sv[PRJ], mx = -3.4e38f;
  #pragma unroll
  for (int v = 0; v < PRJ; ++v) {
    float s = 0.f;
    #pragma unroll
    for (int ks = 0; ks < KSPLIT; ++ks)
      s += sim_part[((size_t)ks * QQ + q) * NPAD + n * PRJ + v];
    sv[v] = s * invt;
    mx = fmaxf(mx, sv[v]);
  }
  float sum = 0.f;
  #pragma unroll
  for (int v = 0; v < PRJ; ++v) { sv[v] = expf(sv[v] - mx); sum += sv[v]; }
  float inv = 1.0f / sum;
  if (t < PRJ) p_w[((size_t)q * N_WAY + n) * PRJ + t] = sv[t] * inv;
  if (t < HW) {
    const float* cs = colsumV + (size_t)n * PRJ * HW + t;
    float a = 0.f;
    #pragma unroll
    for (int v = 0; v < PRJ; ++v) a = fmaf(sv[v] * inv, cs[v * HW], a);
    a *= (1.0f / 64.0f);
    float sg = 1.0f / (1.0f + expf(-a));
    attn_bf[((size_t)n * QQ + q) * KPAD2 + t] = f2bf(sg);
  } else {
    attn_bf[((size_t)n * QQ + q) * KPAD2 + t] = 0;
  }
}

// ---- WMMA GEMM2: async global->LDS staged B, 4 query-tiles share it per block ----
// per n: G[q][896] = attn[q][128] x VPn[128][896]
__global__ __launch_bounds__(128) void g2_gemm(const unsigned short* __restrict__ attn_bf,
                                               const unsigned short* __restrict__ b2p,
                                               float* __restrict__ Gout) {
  // B block for (n, ng): 8 n-tiles x 4 k-tiles x 32 lanes x 16 bf16 = 32 KB
  __shared__ __align__(16) unsigned short lds_b[8 * 4 * 32 * 16];
  const int t = threadIdx.x;
  const int lane = t & 31, wid = t >> 5;
  const int ng = blockIdx.x;   // 0..6  (8 n-tiles each)
  const int qg = blockIdx.y;   // 0..15 (4 q-tiles each)
  const int n  = blockIdx.z;   // 0..4
  const int m = lane & 15, half = lane >> 4;

  // ---- async copy of the shared B block into LDS (ASYNCcnt path) ----
  const char* bsrc = (const char*)(b2p + (size_t)(n * NT2 + ng * 8) * 4 * 32 * 16);
  unsigned lds_base = (unsigned)(unsigned long long)(&lds_b[0]);
  #pragma unroll
  for (int j = 0; j < 16; ++j) {                 // 128 thr * 16 * 16B = 32 KB
    unsigned off = (unsigned)(t + j * 128) * 16u;
    unsigned lds_addr = lds_base + off;
    const void* gp = bsrc + off;
    asm volatile("global_load_async_to_lds_b128 %0, %1, off"
                 :: "v"(lds_addr), "v"(gp) : "memory");
  }
  asm volatile("s_wait_asynccnt 0x0" ::: "memory");
  __syncthreads();

  // ---- each wave computes one 16-query tile against the staged B ----
  const int qt = qg * 4 + wid;
  v8f c[8] = {};
  const unsigned short* arow = attn_bf + ((size_t)n * QQ + qt * 16 + m) * KPAD2;
  #pragma unroll
  for (int kt = 0; kt < 4; ++kt) {
    BF16x16 a;
    const unsigned short* ap = arow + kt * 32 + half * 8;
    a.q[0] = *(const uint4*)(ap);
    a.q[1] = *(const uint4*)(ap + 16);
    #pragma unroll
    for (int j = 0; j < 8; ++j) {
      BF16x16 b;
      const unsigned short* bp = &lds_b[((j * 4 + kt) * 32 + lane) * 16];
      b.q[0] = *(const uint4*)(bp);          // ds_load_b128
      b.q[1] = *(const uint4*)(bp + 8);
      c[j] = __builtin_amdgcn_wmma_f32_16x16x32_bf16(false, a.v, false, b.v,
                                                     (short)0, c[j], false, false);
    }
  }
  #pragma unroll
  for (int j = 0; j < 8; ++j) {
    int col0 = (ng * 8 + j) * 16 + m;
    #pragma unroll
    for (int r = 0; r < 8; ++r) {
      int q = qt * 16 + r + 8 * half;
      Gout[((size_t)n * QQ + q) * COL2 + col0] = c[j][r];
    }
  }
}

// ---------------- per-query tail: cosines, losses, QSA, DeltaGate ----------------
__global__ __launch_bounds__(32) void f_tail(const float* __restrict__ p_w,
                                             const float* __restrict__ Gout,
                                             const float* __restrict__ qvec,
                                             const float* __restrict__ base,
                                             const float* __restrict__ Wq,
                                             const float* __restrict__ bq,
                                             const float* __restrict__ Wk,
                                             const float* __restrict__ bk,
                                             const float* __restrict__ gate,
                                             const int* __restrict__ label,
                                             float* __restrict__ pred,
                                             float* __restrict__ acc) {
  __shared__ float qv_s[DD];
  __shared__ float buf_s[N_WAY * DD];   // proto, then delta
  __shared__ float qp_s[16];
  __shared__ float kp_s[80];
  const int q = blockIdx.x;
  const int lane = threadIdx.x;
  const int d0 = lane, d1 = lane + 32;

  // proto_q[n][d] = sum_v p_w * G
  float proto[N_WAY][2];
  #pragma unroll
  for (int n = 0; n < N_WAY; ++n) {
    float s0 = 0.f, s1 = 0.f;
    const float* g  = Gout + ((size_t)n * QQ + q) * COL2;
    const float* pw = p_w + ((size_t)q * N_WAY + n) * PRJ;
    #pragma unroll
    for (int v = 0; v < PRJ; ++v) {
      float p = pw[v];
      s0 = fmaf(p, g[v * 64 + d0], s0);
      s1 = fmaf(p, g[v * 64 + d1], s1);
    }
    proto[n][0] = s0; proto[n][1] = s1;
  }
  float qv0 = qvec[q * 64 + d0], qv1 = qvec[q * 64 + d1];
  float qn = fmaxf(sqrtf(wave_sum(qv0 * qv0 + qv1 * qv1)), 1e-8f);

  float cosv[N_WAY];
  #pragma unroll
  for (int n = 0; n < N_WAY; ++n) {
    float dt = wave_sum(qv0 * proto[n][0] + qv1 * proto[n][1]);
    float pn = fmaxf(sqrtf(wave_sum(proto[n][0] * proto[n][0] + proto[n][1] * proto[n][1])), 1e-8f);
    cosv[n] = dt / (qn * pn);
  }
  int lbl = label[q];
  float pos = cosv[lbl];
  float sneg = 0.f, mn = 3.0e38f;
  #pragma unroll
  for (int n = 0; n < N_WAY; ++n) if (n != lbl) { sneg += cosv[n]; mn = fminf(mn, cosv[n]); }
  float align_q = fmaxf((sneg - mn) * (1.0f / 3.0f) - pos + 0.3f, 0.0f);

  // QSA mini cross-attention
  qv_s[d0] = qv0; qv_s[d1] = qv1;
  #pragma unroll
  for (int n = 0; n < N_WAY; ++n) { buf_s[n * 64 + d0] = proto[n][0]; buf_s[n * 64 + d1] = proto[n][1]; }
  __syncthreads();
  if (lane < 16) {
    float s = bq[lane];
    for (int d = 0; d < 64; ++d) s = fmaf(qv_s[d], Wq[d * 16 + lane], s);
    qp_s[lane] = s;
  }
  for (int idx = lane; idx < 80; idx += 32) {
    int n = idx >> 4, e = idx & 15;
    float s = bk[e];
    for (int d = 0; d < 64; ++d) s = fmaf(buf_s[n * 64 + d], Wk[d * 16 + e], s);
    kp_s[idx] = s;
  }
  __syncthreads();
  float sc[N_WAY], smax = -3.4e38f;
  #pragma unroll
  for (int n = 0; n < N_WAY; ++n) {
    float s = 0.f;
    #pragma unroll
    for (int e = 0; e < 16; ++e) s = fmaf(qp_s[e], kp_s[n * 16 + e], s);
    sc[n] = s * 0.25f; smax = fmaxf(smax, sc[n]);
  }
  float ssum = 0.f;
  #pragma unroll
  for (int n = 0; n < N_WAY; ++n) { sc[n] = expf(sc[n] - smax); ssum += sc[n]; }

  // mode weights from gate logits
  float g0 = gate[0], g1 = gate[1], g2 = gate[2];
  float gm = fmaxf(g0, fmaxf(g1, g2));
  float e0 = expf(g0 - gm), e1 = expf(g1 - gm), e2 = expf(g2 - gm);
  float es = e0 + e1 + e2;
  float m0 = e0 / es, m1 = e1 / es, m2 = e2 / es;

  // fused + delta
  float fused[N_WAY][2], del[N_WAY][2];
  #pragma unroll
  for (int n = 0; n < N_WAY; ++n) {
    float a2 = sc[n] / ssum;
    fused[n][0] = proto[n][0] * a2;
    fused[n][1] = proto[n][1] * a2;
    del[n][0] = fabsf(fused[n][0] - base[n * 64 + d0]);
    del[n][1] = fabsf(fused[n][1] - base[n * 64 + d1]);
  }
  __syncthreads();
  #pragma unroll
  for (int n = 0; n < N_WAY; ++n) { buf_s[n * 64 + d0] = del[n][0]; buf_s[n * 64 + d1] = del[n][1]; }
  __syncthreads();

  // exact top-k ranking (ties: lower index first, matching lax.top_k)
  float cos2[N_WAY];
  #pragma unroll
  for (int n = 0; n < N_WAY; ++n) {
    int r0 = 0, r1 = 0;
    for (int dp = 0; dp < 64; ++dp) {
      float v = buf_s[n * 64 + dp];
      r0 += (v > del[n][0]) || (v == del[n][0] && dp < d0);
      r1 += (v > del[n][1]) || (v == del[n][1] && dp < d1);
    }
    float w0 = (r0 < 16 ? m0 : 0.f) + (r0 < 32 ? m1 : 0.f) + (r0 < 48 ? m2 : 0.f);
    float w1 = (r1 < 16 ? m0 : 0.f) + (r1 < 32 ? m1 : 0.f) + (r1 < 48 ? m2 : 0.f);
    float gr0 = fused[n][0] * w0, gr1 = fused[n][1] * w1;
    float inv = 1.0f / fmaxf(sqrtf(wave_sum(gr0 * gr0 + gr1 * gr1)), 1e-12f);
    float ga0 = gr0 * inv, ga1 = gr1 * inv;
    float dt = wave_sum(qv0 * ga0 + qv1 * ga1);
    float gn = fmaxf(sqrtf(wave_sum(ga0 * ga0 + ga1 * ga1)), 1e-8f);
    cos2[n] = dt / (qn * gn);
  }

  // pred + CE loss
  float cmax = cos2[0];
  #pragma unroll
  for (int n = 1; n < N_WAY; ++n) cmax = fmaxf(cmax, cos2[n]);
  float pr[N_WAY], ps = 0.f;
  #pragma unroll
  for (int n = 0; n < N_WAY; ++n) { pr[n] = expf(cos2[n] - cmax); ps += pr[n]; }
  if (lane < N_WAY) pred[q * N_WAY + lane] = pr[lane] / ps;

  float lmax = 32.0f * cmax;
  float lse = 0.f;
  #pragma unroll
  for (int n = 0; n < N_WAY; ++n) lse += expf(32.0f * cos2[n] - lmax);
  float logp = 32.0f * cos2[lbl] - lmax - logf(lse);
  if (lane == 0) { atomicAdd(&acc[0], -logp); atomicAdd(&acc[1], align_q); }
}

__global__ void z_finish(const float* __restrict__ acc, float* __restrict__ out) {
  out[QQ * N_WAY] = acc[0] * (1.0f / QQ) + 0.3f * (acc[1] * (1.0f / QQ));
}

// ---------------- launcher ----------------
extern "C" void kernel_launch(void* const* d_in, const int* in_sizes, int n_in,
                              void* d_out, int out_size, void* d_ws, size_t ws_size,
                              hipStream_t stream) {
  const float* feat = (const float*)d_in[0];
  const float* temperature = (const float*)d_in[1];
  const float* gate = (const float*)d_in[2];
  const float* Wq = (const float*)d_in[3];
  const float* bq = (const float*)d_in[4];
  const float* Wk = (const float*)d_in[5];
  const float* bk = (const float*)d_in[6];
  const int* label = (const int*)d_in[7];
  float* out = (float*)d_out;

  char* w = (char*)d_ws;
  size_t off = 0;
  auto alloc = [&](size_t bytes) -> void* {
    void* p = w + off;
    off = (off + bytes + 255) & ~(size_t)255;
    return p;
  };
  float* inv_s    = (float*)alloc((size_t)NSAMP * HW * 4);
  float* vp       = (float*)alloc((size_t)N_WAY * PRJ * DD * HW * 4);
  float* inv_vf   = (float*)alloc(80 * 4);
  float* colsumV  = (float*)alloc((size_t)N_WAY * PRJ * HW * 4);
  float* basep    = (float*)alloc((size_t)N_WAY * DD * 4);
  float* qvec     = (float*)alloc((size_t)QQ * DD * 4);
  float* sim_part = (float*)alloc((size_t)KSPLIT * QQ * NPAD * 4);
  float* p_w      = (float*)alloc((size_t)QQ * N_WAY * PRJ * 4);
  float* acc      = (float*)alloc(8 * 4);
  unsigned short* qbf     = (unsigned short*)alloc((size_t)QQ * FLEN * 2);
  unsigned short* b1p     = (unsigned short*)alloc((size_t)5 * KT1 * 32 * 16 * 2);
  unsigned short* attn_bf = (unsigned short*)alloc((size_t)N_WAY * QQ * KPAD2 * 2);
  unsigned short* b2p     = (unsigned short*)alloc((size_t)N_WAY * NT2 * 4 * 32 * 16 * 2);
  float* Gout     = (float*)alloc((size_t)N_WAY * QQ * COL2 * 4);
  (void)ws_size; (void)in_sizes; (void)n_in; (void)out_size;

  init_acc_kernel<<<1, 32, 0, stream>>>(acc);
  p0_invnorm<<<(NSAMP * HW + 255) / 256, 256, 0, stream>>>(feat, inv_s);
  p1_viewproto<<<(N_WAY * PRJ * DD * HW + 255) / 256, 256, 0, stream>>>(feat, inv_s, vp);
  p2_invvf<<<N_WAY * PRJ, 256, 0, stream>>>(vp, inv_vf);
  p2b_colsum<<<(N_WAY * PRJ * HW + 255) / 256, 256, 0, stream>>>(vp, colsumV);
  p6_base<<<2, 256, 0, stream>>>(vp, basep);
  p3_packB1<<<(5 * KT1 * 32 * 16) / 256, 256, 0, stream>>>(vp, inv_vf, b1p);
  p4_qvec<<<(QQ * DD) / 256, 256, 0, stream>>>(feat, inv_s, qvec);
  p4b_qbf<<<(QQ * FLEN) / 256, 256, 0, stream>>>(feat, inv_s, qbf);
  p5_packB2<<<(N_WAY * NT2 * 4 * 32 * 16) / 256, 256, 0, stream>>>(vp, b2p);
  g1_gemm<<<dim3(QQ / 16, KSPLIT), 32, 0, stream>>>(qbf, b1p, sim_part);
  s1_softmax<<<dim3(QQ, N_WAY), 128, 0, stream>>>(sim_part, colsumV, temperature, p_w, attn_bf);
  g2_gemm<<<dim3(NT2 / 8, QQ / 16 / 4, N_WAY), 128, 0, stream>>>(attn_bf, b2p, Gout);
  f_tail<<<QQ, 32, 0, stream>>>(p_w, Gout, qvec, basep, Wq, bq, Wk, bk, gate, label, out, acc);
  z_finish<<<1, 1, 0, stream>>>(acc, out);
}
